// RelationalMemory_24352464569336
// MI455X (gfx1250) — matrix operand order repaired and checked by reading, not compiled, not run on hardware
//
#include <hip/hip_runtime.h>
#include <hip/hip_bf16.h>
#include <math.h>

typedef __bf16 bf16_t;
typedef __attribute__((ext_vector_type(8)))  __bf16 v8bf;
typedef __attribute__((ext_vector_type(16))) __bf16 v16bf;
typedef __attribute__((ext_vector_type(8)))  float  v8f;

#define RM_BATCH   256
#define RM_SEQ     32
#define RM_IN      1024
#define RM_SLOTS   8
#define RM_MEM     1024
#define RM_S1      9
#define RM_QKV     3072
#define RM_GATES   2048

// ---------------------------------------------------------------------------
// Fragment loaders (wave32, 16-bit operands); this code's contiguous K order, same for A and B.
// A 16x32: lanes 0-15 hold M=lane, elems 0..15 -> K 0..15
//          lanes 16-31 hold M=lane-16, K 16..31.
// B 32x16 (loaded from B^T row-major, contiguous K): lane half kh holds K kh*16..kh*16+15.
// ---------------------------------------------------------------------------
__device__ __forceinline__ v16bf load_fragA(const bf16_t* __restrict__ p, int kh) {
  v8bf a = *(const v8bf*)(p + kh * 16);
  v8bf b = *(const v8bf*)(p + kh * 16 + 8);
  v16bf r;
#pragma unroll
  for (int i = 0; i < 8; ++i) { r[i] = a[i]; r[i + 8] = b[i]; }
  return r;
}
__device__ __forceinline__ v16bf load_fragB(const bf16_t* __restrict__ p, int kh) {
  v8bf a = *(const v8bf*)(p + kh * 16);
  v8bf b = *(const v8bf*)(p + kh * 16 + 8);
  v16bf r;
#pragma unroll
  for (int i = 0; i < 8; ++i) { r[i] = a[i]; r[i + 8] = b[i]; }
  return r;
}

// ---------------------------------------------------------------------------
// bf16 WMMA GEMM: C[M,N] = A[M,K](bf16) * Bt[N,K]^T(bf16) (+bias) (+relu)
// Block: 256 threads = 8 waves, block tile 128x128, wave tile 64x32 (8 WMMA/K-step)
// ---------------------------------------------------------------------------
__global__ __launch_bounds__(256)
void wmma_gemm_bf16(const bf16_t* __restrict__ A, const bf16_t* __restrict__ Bt,
                    const float* __restrict__ bias,
                    float* __restrict__ Cf, bf16_t* __restrict__ Cb,
                    int M, int N, int K, int relu)
{
  const int lane = threadIdx.x & 31;
  const int wave = threadIdx.x >> 5;
  const int r    = lane & 15;
  const int kh   = lane >> 4;
  const int rowBase = blockIdx.y * 128 + (wave >> 2) * 64;
  const int colBase = blockIdx.x * 128 + (wave & 3) * 32;

  v8f acc[4][2];
#pragma unroll
  for (int i = 0; i < 4; ++i)
#pragma unroll
    for (int j = 0; j < 2; ++j) acc[i][j] = {};

  const bf16_t* aP[4];
  const bf16_t* bP[2];
#pragma unroll
  for (int i = 0; i < 4; ++i) aP[i] = A + (size_t)(rowBase + i * 16 + r) * K;
#pragma unroll
  for (int j = 0; j < 2; ++j) bP[j] = Bt + (size_t)(colBase + j * 16 + r) * K;

  for (int k = 0; k < K; k += 32) {
    if (k + 256 < K) {  // L2-resident operands: keep the L0/WGP$ primed
#pragma unroll
      for (int i = 0; i < 4; ++i) __builtin_prefetch(aP[i] + k + 256, 0, 1);
#pragma unroll
      for (int j = 0; j < 2; ++j) __builtin_prefetch(bP[j] + k + 256, 0, 1);
    }
    v16bf af[4];
#pragma unroll
    for (int i = 0; i < 4; ++i) af[i] = load_fragA(aP[i] + k, kh);
    v16bf bfrag[2];
#pragma unroll
    for (int j = 0; j < 2; ++j) bfrag[j] = load_fragB(bP[j] + k, kh);
#pragma unroll
    for (int i = 0; i < 4; ++i)
#pragma unroll
      for (int j = 0; j < 2; ++j)
        acc[i][j] = __builtin_amdgcn_wmma_f32_16x16x32_bf16(
            false, af[i], false, bfrag[j], (short)0, acc[i][j], false, false);
  }

  // C/D layout: VGPR e holds (M = (lane>>4)*8 + e, N = lane&15) within each 16x16 tile
#pragma unroll
  for (int i = 0; i < 4; ++i) {
#pragma unroll
    for (int j = 0; j < 2; ++j) {
      const int col = colBase + j * 16 + r;
      const float bv = bias ? bias[col] : 0.0f;
#pragma unroll
      for (int e = 0; e < 8; ++e) {
        const int row = rowBase + i * 16 + kh * 8 + e;
        float v = acc[i][j][e] + bv;
        if (relu) v = fmaxf(v, 0.0f);
        const size_t o = (size_t)row * N + col;
        if (Cf) Cf[o] = v;
        if (Cb) Cb[o] = (bf16_t)v;
      }
    }
  }
}

// ---------------------------------------------------------------------------
// _ln2: layernorm over the whole (S1, D) block per batch element.
// One block per batch element; n = S1*D; g/b are flat (S1*D) and indexed by j.
// Optionally adds a residual stream and emits a bf16 copy for the next GEMM.
// ---------------------------------------------------------------------------
__global__ __launch_bounds__(256)
void ln2_kernel(const float* __restrict__ x, const float* __restrict__ addn,
                const float* __restrict__ g, const float* __restrict__ bb,
                float* __restrict__ outf, bf16_t* __restrict__ outb, int n)
{
  const int b = blockIdx.x;
  const float* xb = x + (size_t)b * n;
  const float* ab = addn ? addn + (size_t)b * n : nullptr;
  float s = 0.0f, s2 = 0.0f;
  for (int j = threadIdx.x; j < n; j += 256) {
    float v = xb[j] + (ab ? ab[j] : 0.0f);
    s += v; s2 += v * v;
  }
  __shared__ float sh[256], sh2[256];
  sh[threadIdx.x] = s; sh2[threadIdx.x] = s2;
  __syncthreads();
  for (int st = 128; st > 0; st >>= 1) {
    if (threadIdx.x < st) { sh[threadIdx.x] += sh[threadIdx.x + st]; sh2[threadIdx.x] += sh2[threadIdx.x + st]; }
    __syncthreads();
  }
  const float mean = sh[0] / (float)n;
  const float var  = sh2[0] / (float)n - mean * mean;
  const float rs   = rsqrtf(var + 1e-5f);
  for (int j = threadIdx.x; j < n; j += 256) {
    float v = xb[j] + (ab ? ab[j] : 0.0f);
    float y = (v - mean) * rs * g[j] + bb[j];
    outf[(size_t)b * n + j] = y;
    if (outb) outb[(size_t)b * n + j] = (bf16_t)y;
  }
}

// ---------------------------------------------------------------------------
// Attention over S1=9 slots, one block per (batch, head), 128 threads.
// qkv layout: [B*S1, 3072], head h occupies cols h*384 + {q:0..127,k:128..255,v:256..383}
// ---------------------------------------------------------------------------
__global__ __launch_bounds__(128)
void attn_kernel(const float* __restrict__ qkv, float* __restrict__ outp)
{
  const int bh = blockIdx.x;
  const int b = bh >> 3, h = bh & 7;
  __shared__ float q[9][128], k[9][128], v[9][128], att[9][12];
  const float scale = 0.08838834764831845f;  // 128^-0.5
  for (int idx = threadIdx.x; idx < 9 * 128; idx += 128) {
    const int s = idx >> 7, d = idx & 127;
    const float* base = qkv + (size_t)(b * 9 + s) * RM_QKV + h * 384;
    q[s][d] = base[d] * scale;
    k[s][d] = base[128 + d];
    v[s][d] = base[256 + d];
  }
  __syncthreads();
  if (threadIdx.x < 81) {
    const int i = threadIdx.x / 9, j = threadIdx.x % 9;
    float a = 0.0f;
#pragma unroll 16
    for (int d = 0; d < 128; ++d) a += q[i][d] * k[j][d];
    att[i][j] = a;
  }
  __syncthreads();
  if (threadIdx.x < 9) {
    const int i = threadIdx.x;
    float mx = -1e30f;
    for (int j = 0; j < 9; ++j) mx = fmaxf(mx, att[i][j]);
    float sm = 0.0f;
    for (int j = 0; j < 9; ++j) { float e = __expf(att[i][j] - mx); att[i][j] = e; sm += e; }
    const float inv = 1.0f / sm;
    for (int j = 0; j < 9; ++j) att[i][j] *= inv;
  }
  __syncthreads();
  const int d = threadIdx.x;
  for (int i = 0; i < 9; ++i) {
    float a = 0.0f;
    for (int j = 0; j < 9; ++j) a += att[i][j] * v[j][d];
    outp[(size_t)(b * 9 + i) * RM_MEM + h * 128 + d] = a;
  }
}

// ---------------------------------------------------------------------------
// Elementwise / packing kernels
// ---------------------------------------------------------------------------
__global__ void transpose_cast_kernel(const float* __restrict__ W, bf16_t* __restrict__ Wt,
                                      int K, int N)  // W[K,N] -> Wt[N,K] bf16
{
  const size_t idx = (size_t)blockIdx.x * blockDim.x + threadIdx.x;
  if (idx >= (size_t)K * N) return;
  const int n = (int)(idx / K), k = (int)(idx % K);
  Wt[idx] = (bf16_t)W[(size_t)k * N + n];
}

__global__ void cast_bf_kernel(const float* __restrict__ x, bf16_t* __restrict__ y, size_t n)
{
  const size_t idx = (size_t)blockIdx.x * blockDim.x + threadIdx.x;
  if (idx < n) y[idx] = (bf16_t)x[idx];
}

__global__ void tanh_cast_kernel(const float* __restrict__ x, bf16_t* __restrict__ y, size_t n)
{
  const size_t idx = (size_t)blockIdx.x * blockDim.x + threadIdx.x;
  if (idx < n) y[idx] = (bf16_t)tanhf(x[idx]);
}

// build m = concat(mem, inp_t) : rows b*9+s; s<8 -> mem[b*8+s], s==8 -> inp_all[b*SEQ+t]
__global__ void pack_m_kernel(const float* __restrict__ memc, const float* __restrict__ inp_all,
                              float* __restrict__ mf, bf16_t* __restrict__ mb, int t)
{
  const size_t idx = (size_t)blockIdx.x * blockDim.x + threadIdx.x;
  if (idx >= (size_t)RM_BATCH * RM_S1 * RM_MEM) return;
  const int d = (int)(idx & (RM_MEM - 1));
  const int row = (int)(idx >> 10);
  const int b = row / RM_S1, s = row - b * RM_S1;
  float v;
  if (s < RM_SLOTS) v = memc[((size_t)b * RM_SLOTS + s) * RM_MEM + d];
  else              v = inp_all[((size_t)b * RM_SEQ + t) * RM_MEM + d];
  mf[idx] = v;
  mb[idx] = (bf16_t)v;
}

__device__ __forceinline__ float sigmoidf_(float x) { return 1.0f / (1.0f + __expf(-x)); }

// gates = mg_out + igb_all[b,t]; next = sig(ig+ib)*tanh(m_pre) + sig(fg+fb)*mem
__global__ void gate_final_kernel(const float* __restrict__ mg_out, const float* __restrict__ igb_all,
                                  const float* __restrict__ mpre, const float* __restrict__ memc,
                                  float* __restrict__ out_rows,
                                  const float* __restrict__ fb_p, const float* __restrict__ ib_p, int t)
{
  const size_t idx = (size_t)blockIdx.x * blockDim.x + threadIdx.x;
  if (idx >= (size_t)RM_BATCH * RM_SLOTS * RM_MEM) return;
  const int d = (int)(idx & (RM_MEM - 1));
  const int bs = (int)(idx >> 10);
  const int b = bs >> 3, s = bs & 7;
  const size_t grow = (size_t)bs * RM_GATES;
  const size_t irow = ((size_t)b * RM_SEQ + t) * RM_GATES;
  const float ig = mg_out[grow + d]            + igb_all[irow + d]            + ib_p[0];
  const float fg = mg_out[grow + RM_MEM + d]   + igb_all[irow + RM_MEM + d]   + fb_p[0];
  const float pre = mpre[((size_t)b * RM_S1 + s) * RM_MEM + d];
  const float mo  = memc[(size_t)bs * RM_MEM + d];
  out_rows[(size_t)b * (RM_SLOTS * RM_MEM) + s * RM_MEM + d] =
      sigmoidf_(ig) * tanhf(pre) + sigmoidf_(fg) * mo;
}

// ---------------------------------------------------------------------------
// Host orchestration
// ---------------------------------------------------------------------------
static inline unsigned ceil_div_u(size_t n, unsigned b) { return (unsigned)((n + b - 1) / b); }

extern "C" void kernel_launch(void* const* d_in, const int* in_sizes, int n_in,
                              void* d_out, int out_size, void* d_ws, size_t ws_size,
                              hipStream_t stream)
{
  const float* inputs  = (const float*)d_in[0];
  const float* memory0 = (const float*)d_in[1];
  const float* W_qkv   = (const float*)d_in[2];
  const float* b_qkv   = (const float*)d_in[3];
  const float* g_qln   = (const float*)d_in[4];
  const float* b_qln   = (const float*)d_in[5];
  const float* W_mlp   = (const float*)d_in[6];
  const float* b_mlp   = (const float*)d_in[7];
  const float* g_ln1   = (const float*)d_in[8];
  const float* b_ln1   = (const float*)d_in[9];
  const float* g_ln2   = (const float*)d_in[10];
  const float* b_ln2   = (const float*)d_in[11];
  const float* W_in    = (const float*)d_in[12];
  const float* b_in    = (const float*)d_in[13];
  const float* W_ig    = (const float*)d_in[14];
  const float* b_ig    = (const float*)d_in[15];
  const float* W_mg    = (const float*)d_in[16];
  const float* b_mg    = (const float*)d_in[17];
  const float* fb_p    = (const float*)d_in[18];
  const float* ib_p    = (const float*)d_in[19];
  float* dout = (float*)d_out;

  // ---- workspace carve-up ----
  char* base = (char*)d_ws;
  size_t off = 0;
  auto carve = [&](size_t bytes) -> char* {
    char* p = base + off;
    off += (bytes + 255) & ~(size_t)255;
    return p;
  };
  const size_t BT = (size_t)RM_BATCH * RM_SEQ;          // 8192 rows
  bf16_t* WinT  = (bf16_t*)carve((size_t)RM_MEM * RM_IN * 2);
  bf16_t* WqkvT = (bf16_t*)carve((size_t)RM_QKV * RM_MEM * 2);
  bf16_t* WmlpT = (bf16_t*)carve((size_t)RM_MEM * RM_MEM * 2);
  bf16_t* WigT  = (bf16_t*)carve((size_t)RM_GATES * RM_MEM * 2);
  bf16_t* WmgT  = (bf16_t*)carve((size_t)RM_GATES * RM_MEM * 2);
  bf16_t* x_bf  = (bf16_t*)carve(BT * RM_IN * 2);
  float*  inp_f = (float*) carve(BT * RM_MEM * 4);
  bf16_t* inp_b = (bf16_t*)carve(BT * RM_MEM * 2);
  float*  igb   = (float*) carve(BT * RM_GATES * 4);
  float*  m_f   = (float*) carve((size_t)RM_BATCH * RM_S1 * RM_MEM * 4);
  bf16_t* m_b   = (bf16_t*)carve((size_t)RM_BATCH * RM_S1 * RM_MEM * 2);
  float*  qkv   = (float*) carve((size_t)RM_BATCH * RM_S1 * RM_QKV * 4);
  float*  att_o = (float*) carve((size_t)RM_BATCH * RM_S1 * RM_MEM * 4);
  bf16_t* h1_b  = (bf16_t*)carve((size_t)RM_BATCH * RM_S1 * RM_MEM * 2);
  float*  h2_f  = (float*) carve((size_t)RM_BATCH * RM_S1 * RM_MEM * 4);
  bf16_t* th_b  = (bf16_t*)carve((size_t)RM_BATCH * RM_SLOTS * RM_MEM * 2);
  float*  mg_o  = (float*) carve((size_t)RM_BATCH * RM_SLOTS * RM_GATES * 4);
  (void)ws_size; (void)n_in; (void)in_sizes; (void)out_size;

  // ---- prep: weights -> bf16 [N,K]; hoisted input GEMMs for all timesteps ----
  transpose_cast_kernel<<<ceil_div_u((size_t)RM_IN * RM_MEM, 256), 256, 0, stream>>>(W_in,  WinT,  RM_IN,  RM_MEM);
  transpose_cast_kernel<<<ceil_div_u((size_t)RM_MEM * RM_QKV, 256), 256, 0, stream>>>(W_qkv, WqkvT, RM_MEM, RM_QKV);
  transpose_cast_kernel<<<ceil_div_u((size_t)RM_MEM * RM_MEM, 256), 256, 0, stream>>>(W_mlp, WmlpT, RM_MEM, RM_MEM);
  transpose_cast_kernel<<<ceil_div_u((size_t)RM_MEM * RM_GATES, 256), 256, 0, stream>>>(W_ig, WigT, RM_MEM, RM_GATES);
  transpose_cast_kernel<<<ceil_div_u((size_t)RM_MEM * RM_GATES, 256), 256, 0, stream>>>(W_mg, WmgT, RM_MEM, RM_GATES);
  cast_bf_kernel<<<ceil_div_u(BT * RM_IN, 256), 256, 0, stream>>>(inputs, x_bf, BT * RM_IN);

  // inp_all = x @ W_in + b_in   [8192,1024]
  {
    dim3 g(RM_MEM / 128, (unsigned)(BT / 128));
    wmma_gemm_bf16<<<g, 256, 0, stream>>>(x_bf, WinT, b_in, inp_f, inp_b,
                                          (int)BT, RM_MEM, RM_IN, 0);
  }
  // igb_all = inp @ W_ig + b_ig [8192,2048]
  {
    dim3 g(RM_GATES / 128, (unsigned)(BT / 128));
    wmma_gemm_bf16<<<g, 256, 0, stream>>>(inp_b, WigT, b_ig, igb, nullptr,
                                          (int)BT, RM_GATES, RM_MEM, 0);
  }

  const size_t rows_per_t = (size_t)RM_BATCH * RM_SLOTS * RM_MEM;  // 2,097,152 floats
  const int MR = RM_BATCH * RM_S1;  // 2304, divisible by 128

  for (int t = 0; t < RM_SEQ; ++t) {
    const float* memc = (t == 0) ? memory0 : (dout + (size_t)(t - 1) * rows_per_t);
    float* out_rows = dout + (size_t)t * rows_per_t;

    pack_m_kernel<<<ceil_div_u((size_t)MR * RM_MEM, 256), 256, 0, stream>>>(memc, inp_f, m_f, m_b, t);

    // qkv = m @ W_qkv + b_qkv, then _ln2 over (S1, 3072) per batch
    {
      dim3 g(RM_QKV / 128, MR / 128);
      wmma_gemm_bf16<<<g, 256, 0, stream>>>(m_b, WqkvT, b_qkv, qkv, nullptr, MR, RM_QKV, RM_MEM, 0);
    }
    ln2_kernel<<<RM_BATCH, 256, 0, stream>>>(qkv, nullptr, g_qln, b_qln, qkv, nullptr, RM_S1 * RM_QKV);

    attn_kernel<<<RM_BATCH * 8, 128, 0, stream>>>(qkv, att_o);

    // m = ln1(m + attn_out)  (also emit bf16 for MLP)
    ln2_kernel<<<RM_BATCH, 256, 0, stream>>>(m_f, att_o, g_ln1, b_ln1, m_f, m_b, RM_S1 * RM_MEM);

    // MLP x2 with relu
    {
      dim3 g(RM_MEM / 128, MR / 128);
      wmma_gemm_bf16<<<g, 256, 0, stream>>>(m_b, WmlpT, b_mlp, nullptr, h1_b, MR, RM_MEM, RM_MEM, 1);
      wmma_gemm_bf16<<<g, 256, 0, stream>>>(h1_b, WmlpT, b_mlp, h2_f, nullptr, MR, RM_MEM, RM_MEM, 1);
    }
    // m = ln2(m + h)
    ln2_kernel<<<RM_BATCH, 256, 0, stream>>>(m_f, h2_f, g_ln2, b_ln2, m_f, nullptr, RM_S1 * RM_MEM);

    // gates = tanh(mem) @ W_mg + b_mg  [2048,2048]
    tanh_cast_kernel<<<ceil_div_u((size_t)RM_BATCH * RM_SLOTS * RM_MEM, 256), 256, 0, stream>>>(
        memc, th_b, (size_t)RM_BATCH * RM_SLOTS * RM_MEM);
    {
      dim3 g(RM_GATES / 128, (RM_BATCH * RM_SLOTS) / 128);
      wmma_gemm_bf16<<<g, 256, 0, stream>>>(th_b, WmgT, b_mg, mg_o, nullptr,
                                            RM_BATCH * RM_SLOTS, RM_GATES, RM_MEM, 0);
    }
    gate_final_kernel<<<ceil_div_u((size_t)RM_BATCH * RM_SLOTS * RM_MEM, 256), 256, 0, stream>>>(
        mg_o, igb, m_f, memc, out_rows, fb_p, ib_p, t);
  }

  // final_mem = last step's next_mem, appended after the logits block
  const size_t logits_elems = (size_t)RM_BATCH * RM_SEQ * RM_SLOTS * RM_MEM;  // 67,108,864
  hipMemcpyAsync(dout + logits_elems, dout + (size_t)(RM_SEQ - 1) * rows_per_t,
                 rows_per_t * sizeof(float), hipMemcpyDeviceToDevice, stream);
}
